// FusedMoE_76974403879011
// MI455X (gfx1250) — matrix operand order repaired and checked by reading, not compiled, not run on hardware
//
#include <hip/hip_runtime.h>

#define T_TOK 1024
#define H_DIM 1024
#define E_NUM 8
#define I_DIM 2048
#define TOPK  2
#define P_MAX 2560              // 2048 pairs + up to 8*63 -> padded to 64/expert
#define MAX_BLOCKS (P_MAX / 64) // 40 row-blocks of 64 rows (4 M-tiles)

typedef __attribute__((ext_vector_type(16))) __bf16          v16bf;
typedef __attribute__((ext_vector_type(16))) unsigned short  v16u;
typedef __attribute__((ext_vector_type(8)))  float           v8f;
typedef __attribute__((ext_vector_type(8)))  unsigned short  v8u;

__device__ __forceinline__ unsigned short f2bf(float f) {
  __bf16 h = (__bf16)f;                       // RNE fptrunc; native v_cvt on gfx1250
  return __builtin_bit_cast(unsigned short, h);
}

// 16 contiguous fp32 -> v16bf (B fragment fallback path)
__device__ __forceinline__ v16bf load16f_bf(const float* p) {
  const float4* q = (const float4*)p;
  float4 f0 = q[0], f1 = q[1], f2 = q[2], f3 = q[3];
  v16u u;
  u[0]=f2bf(f0.x); u[1]=f2bf(f0.y); u[2]=f2bf(f0.z); u[3]=f2bf(f0.w);
  u[4]=f2bf(f1.x); u[5]=f2bf(f1.y); u[6]=f2bf(f1.z); u[7]=f2bf(f1.w);
  u[8]=f2bf(f2.x); u[9]=f2bf(f2.y); u[10]=f2bf(f2.z); u[11]=f2bf(f2.w);
  u[12]=f2bf(f3.x); u[13]=f2bf(f3.y); u[14]=f2bf(f3.z); u[15]=f2bf(f3.w);
  return __builtin_bit_cast(v16bf, u);
}
// 16 contiguous bf16 -> v16bf (B fragment, pre-converted path)
__device__ __forceinline__ v16bf loadB_bf(const unsigned short* p) {
  return __builtin_bit_cast(v16bf, *(const v16u*)p);
}
// A fragment: K = {kb..kb+15} contiguous; p already offset by half*16
__device__ __forceinline__ v16bf loadA_bf(const unsigned short* p) {
  v8u a0 = *(const v8u*)p;
  v8u a1 = *(const v8u*)(p + 8);
  v16u u;
#pragma unroll
  for (int j = 0; j < 8; j++) { u[j] = a0[j]; u[8 + j] = a1[j]; }
  return __builtin_bit_cast(v16bf, u);
}

// ---------------- fp32 -> bf16 bulk converter (8 elems/thread) ----------------
__global__ __launch_bounds__(256)
void cvt_f32_bf16(const float* __restrict__ src, unsigned short* __restrict__ dst, int n) {
  int i = (blockIdx.x * 256 + threadIdx.x) * 8;
  if (i >= n) return;
  float4 f0 = *(const float4*)(src + i);
  float4 f1 = *(const float4*)(src + i + 4);
  v8u o;
  o[0]=f2bf(f0.x); o[1]=f2bf(f0.y); o[2]=f2bf(f0.z); o[3]=f2bf(f0.w);
  o[4]=f2bf(f1.x); o[5]=f2bf(f1.y); o[6]=f2bf(f1.z); o[7]=f2bf(f1.w);
  *(v8u*)(dst + i) = o;
}

// ---------------- Kernel 1: router logits + softmax top-2 ----------------
__global__ __launch_bounds__(256)
void moe_router(const float* __restrict__ x, const float* __restrict__ gw,
                float* __restrict__ logits_out, int* __restrict__ topk_idx,
                float* __restrict__ topk_w) {
  int t    = blockIdx.x;
  int e    = threadIdx.x >> 5;
  int lane = threadIdx.x & 31;
  const float* xr = x  + (size_t)t * H_DIM;
  const float* gr = gw + (size_t)e * H_DIM;
  float s = 0.f;
  for (int h = lane; h < H_DIM; h += 32) s += xr[h] * gr[h];
  for (int off = 16; off > 0; off >>= 1) s += __shfl_down(s, off, 32);
  __shared__ float lg[E_NUM];
  if (lane == 0) lg[e] = s;
  __syncthreads();
  if (threadIdx.x == 0) {
    for (int i = 0; i < E_NUM; i++) logits_out[(size_t)t * E_NUM + i] = lg[i];
    int i0 = 0;
    for (int i = 1; i < E_NUM; i++) if (lg[i] > lg[i0]) i0 = i;
    int i1 = (i0 == 0) ? 1 : 0;
    for (int i = 0; i < E_NUM; i++) if (i != i0 && lg[i] > lg[i1]) i1 = i;
    // renormalized top-2 weights: p1/(p0+p1) = 1/(1+exp(l0-l1))
    float w1 = __builtin_amdgcn_rcpf(1.0f + __expf(lg[i0] - lg[i1]));
    float w0 = 1.0f - w1;
    topk_idx[t * 2 + 0] = i0; topk_idx[t * 2 + 1] = i1;
    topk_w[t * 2 + 0]   = w0; topk_w[t * 2 + 1]   = w1;
  }
}

// ---------------- Kernel 2: dispatch lists, LDS-staged + lane-parallel fill ----------------
__global__ __launch_bounds__(256)
void moe_dispatch(const int* __restrict__ topk_idx,
                  int* __restrict__ pair_token,
                  int* __restrict__ pair_pos,
                  int* __restrict__ blk_expert,
                  int* __restrict__ n_blocks) {
  __shared__ int sidx[T_TOK * TOPK];     // 8 KB staged expert ids
  __shared__ int scnt[E_NUM];
  __shared__ int soffs[E_NUM + 1];
  int tid = threadIdx.x;
  if (tid < E_NUM) scnt[tid] = 0;
  __syncthreads();
  int base = tid * 8;                    // 256 threads x 8 = 2048
#pragma unroll
  for (int j = 0; j < 8; j++) {
    int v = topk_idx[base + j];
    sidx[base + j] = v;
    atomicAdd(&scnt[v], 1);              // ds_add; order-independent -> deterministic
  }
  __syncthreads();
  if (tid == 0) {
    soffs[0] = 0;
    for (int e = 0; e < E_NUM; e++) soffs[e + 1] = soffs[e] + ((scnt[e] + 63) & ~63);
    *n_blocks = soffs[E_NUM] >> 6;
  }
  __syncthreads();
  if (tid < E_NUM) {                     // lane e fills expert e's segment (stable order)
    int e = tid;
    int cur = soffs[e];
    for (int p = 0; p < T_TOK * TOPK; p++) {
      if (sidx[p] == e) {
        pair_token[cur] = p >> 1;        // token id (TOPK == 2)
        pair_pos[p] = cur;
        cur++;
      }
    }
    for (int p = cur; p < soffs[e + 1]; p++) pair_token[p] = 0;   // dummy rows
  }
  if (tid < MAX_BLOCKS) {                // per-row-block owning expert
    int r = tid * 64;
    int e = 0;
#pragma unroll
    for (int k = 0; k < E_NUM; k++) if (r >= soffs[k + 1]) e = k + 1;
    blk_expert[tid] = (e < E_NUM) ? e : (E_NUM - 1);   // blocks >= n_blocks unused
  }
}

// ---------------- Kernel 3: act = silu(x@w1^T)*(x@w3^T); 64 rows x 64 I-cols/block ----------------
template<bool PRE>
__global__ __launch_bounds__(128)
void moe_gemm1(const unsigned short* __restrict__ xbf,
               const float* __restrict__ w13,
               const unsigned short* __restrict__ w13bf,
               const int* __restrict__ pair_token,
               const int* __restrict__ blk_expert,
               const int* __restrict__ n_blocks,
               unsigned short* __restrict__ act) {
  int blk = blockIdx.x;
  if (blk >= *n_blocks) return;                 // block-uniform: EXEC all-1 for WMMA
  int e     = blk_expert[blk];
  int wave  = threadIdx.x >> 5;
  int lane  = threadIdx.x & 31;
  int half  = lane >> 4;
  int l16   = lane & 15;
  int i0    = blockIdx.y * 64 + wave * 16;
  int rbase = blk * 64;

  const unsigned short* ax[4];
#pragma unroll
  for (int mt = 0; mt < 4; mt++) {
    int tok = pair_token[rbase + mt * 16 + l16];
    ax[mt] = xbf + (size_t)tok * H_DIM + half * 16;
  }
  size_t rowg = ((size_t)e * (2 * I_DIM) + (i0 + l16)) * H_DIM + half * 16;
  size_t rowu = rowg + (size_t)I_DIM * H_DIM;

  v8f accg[4] = {{}, {}, {}, {}};
  v8f accu[4] = {{}, {}, {}, {}};
#pragma unroll 2
  for (int k0 = 0; k0 < H_DIM; k0 += 32) {
    v16bf b1, b3;
    if constexpr (PRE) {
      b1 = loadB_bf(w13bf + rowg + k0);
      b3 = loadB_bf(w13bf + rowu + k0);
    } else {
      b1 = load16f_bf(w13 + rowg + k0);
      b3 = load16f_bf(w13 + rowu + k0);
    }
#pragma unroll
    for (int mt = 0; mt < 4; mt++) {
      v16bf a = loadA_bf(ax[mt] + k0);
      accg[mt] = __builtin_amdgcn_wmma_f32_16x16x32_bf16(false, a, false, b1, (short)0, accg[mt], false, false);
      accu[mt] = __builtin_amdgcn_wmma_f32_16x16x32_bf16(false, a, false, b3, (short)0, accu[mt], false, false);
    }
  }
#pragma unroll
  for (int mt = 0; mt < 4; mt++) {
    unsigned short* ocol = act + (size_t)(rbase + mt * 16 + half * 8) * I_DIM + (i0 + l16);
#pragma unroll
    for (int j = 0; j < 8; j++) {
      float g = accg[mt][j];
      float v = (g * __builtin_amdgcn_rcpf(1.0f + __expf(-g))) * accu[mt][j]; // silu(g)*u
      ocol[(size_t)j * I_DIM] = f2bf(v);
    }
  }
}

// ---------------- Kernel 4: expert_out = act @ w2[e]^T; 64 rows x 64 H-cols/block ----------------
template<bool PRE>
__global__ __launch_bounds__(128)
void moe_gemm2(const unsigned short* __restrict__ act,
               const float* __restrict__ w2,
               const unsigned short* __restrict__ w2bf,
               const int* __restrict__ blk_expert,
               const int* __restrict__ n_blocks,
               float* __restrict__ eo) {
  int blk = blockIdx.x;
  if (blk >= *n_blocks) return;
  int e     = blk_expert[blk];
  int wave  = threadIdx.x >> 5;
  int lane  = threadIdx.x & 31;
  int half  = lane >> 4;
  int l16   = lane & 15;
  int h0    = blockIdx.y * 64 + wave * 16;
  int rbase = blk * 64;

  const unsigned short* arow = act + (size_t)(rbase + l16) * I_DIM + half * 16;
  size_t browo = ((size_t)e * H_DIM + (h0 + l16)) * I_DIM + half * 16;

  v8f acc[4] = {{}, {}, {}, {}};
#pragma unroll 2
  for (int k0 = 0; k0 < I_DIM; k0 += 32) {
    v16bf b;
    if constexpr (PRE) b = loadB_bf(w2bf + browo + k0);
    else               b = load16f_bf(w2 + browo + k0);
#pragma unroll
    for (int mt = 0; mt < 4; mt++) {
      v16bf a = loadA_bf(arow + (size_t)mt * 16 * I_DIM + k0);
      acc[mt] = __builtin_amdgcn_wmma_f32_16x16x32_bf16(false, a, false, b, (short)0, acc[mt], false, false);
    }
  }
#pragma unroll
  for (int mt = 0; mt < 4; mt++) {
    float* ocol = eo + (size_t)(rbase + mt * 16 + half * 8) * H_DIM + (h0 + l16);
#pragma unroll
    for (int j = 0; j < 8; j++) ocol[(size_t)j * H_DIM] = acc[mt][j];
  }
}

// ---------------- Kernel 5: gather-combine (no atomics, deterministic) ----------------
__global__ __launch_bounds__(256)
void moe_combine(const float* __restrict__ eo, const int* __restrict__ pair_pos,
                 const float* __restrict__ topk_w, float* __restrict__ out) {
  int idx = blockIdx.x * 256 + threadIdx.x;
  if (idx >= T_TOK * H_DIM) return;
  int t = idx >> 10;
  int h = idx & (H_DIM - 1);
  int p0 = pair_pos[t * 2 + 0], p1 = pair_pos[t * 2 + 1];
  float w0 = topk_w[t * 2 + 0], w1 = topk_w[t * 2 + 1];
  out[idx] = w0 * eo[(size_t)p0 * H_DIM + h] + w1 * eo[(size_t)p1 * H_DIM + h];
}

extern "C" void kernel_launch(void* const* d_in, const int* in_sizes, int n_in,
                              void* d_out, int out_size, void* d_ws, size_t ws_size,
                              hipStream_t stream) {
  const float* x   = (const float*)d_in[0];   // [T, H]
  const float* gw  = (const float*)d_in[1];   // [E, H]
  const float* w13 = (const float*)d_in[2];   // [E, 2I, H]
  const float* w2  = (const float*)d_in[3];   // [E, H, I]
  float* out        = (float*)d_out;                      // [T, H]
  float* logits_out = out + (size_t)T_TOK * H_DIM;        // [T, E]

  char* ws = (char*)d_ws;
  int*   topk_idx   = (int*)(ws + 0);          // 2048 ints
  float* topk_w     = (float*)(ws + 8192);     // 2048 floats
  int*   pair_pos   = (int*)(ws + 16384);      // 2048 ints
  int*   pair_token = (int*)(ws + 24576);      // P_MAX ints (ends 34816)
  int*   blk_expert = (int*)(ws + 34816);      // 40 ints
  int*   n_blocks   = (int*)(ws + 35072);      // 1 int
  unsigned short* xbf = (unsigned short*)(ws + 36864);                 // [T,H] bf16 (2 MB)
  unsigned short* act = (unsigned short*)(ws + 2134016);               // [P_MAX,I] bf16 (10 MB)
  float* eo           = (float*)(ws + 12619776);                       // [P_MAX,H] f32 (10 MB)
  unsigned short* w13bf = (unsigned short*)(ws + 23105536);            // 67 MB
  unsigned short* w2bf  = (unsigned short*)(ws + 90214400);            // 33.5 MB
  const size_t WS_NEED_PRE = 123768832ull;
  bool preconv = (ws_size >= WS_NEED_PRE);     // constant across calls -> deterministic

  moe_router<<<T_TOK, 256, 0, stream>>>(x, gw, logits_out, topk_idx, topk_w);
  moe_dispatch<<<1, 256, 0, stream>>>(topk_idx, pair_token, pair_pos, blk_expert, n_blocks);

  // x -> bf16 always (2 MB)
  cvt_f32_bf16<<<(T_TOK * H_DIM) / 2048, 256, 0, stream>>>(x, xbf, T_TOK * H_DIM);

  dim3 g1(MAX_BLOCKS, I_DIM / 64);
  dim3 g2(MAX_BLOCKS, H_DIM / 64);
  if (preconv) {
    const int n13 = E_NUM * 2 * I_DIM * H_DIM;   // 33,554,432
    const int n2  = E_NUM * H_DIM * I_DIM;       // 16,777,216
    cvt_f32_bf16<<<n13 / 2048, 256, 0, stream>>>(w13, w13bf, n13);
    cvt_f32_bf16<<<n2 / 2048, 256, 0, stream>>>(w2, w2bf, n2);
    moe_gemm1<true><<<g1, 128, 0, stream>>>(xbf, w13, w13bf, pair_token, blk_expert, n_blocks, act);
    moe_gemm2<true><<<g2, 128, 0, stream>>>(act, w2, w2bf, blk_expert, n_blocks, eo);
  } else {
    moe_gemm1<false><<<g1, 128, 0, stream>>>(xbf, w13, w13bf, pair_token, blk_expert, n_blocks, act);
    moe_gemm2<false><<<g2, 128, 0, stream>>>(act, w2, w2bf, blk_expert, n_blocks, eo);
  }
  moe_combine<<<(T_TOK * H_DIM + 255) / 256, 256, 0, stream>>>(eo, pair_pos, topk_w, out);
}
